// NJXAttention_14894946582534
// MI455X (gfx1250) — matrix operand order repaired and checked by reading, not compiled, not run on hardware
//
#include <hip/hip_runtime.h>
#include <hip/hip_bf16.h>
#include <cstdint>

// ---------------------------------------------------------------------------
// Problem constants (B, T, D, H, W) = (4, 4096, 1024, 16, 64)
// ---------------------------------------------------------------------------
static constexpr int kB  = 4;
static constexpr int kT  = 4096;
static constexpr int kD  = 1024;
static constexpr int kH  = 16;
static constexpr int kWin = 64;     // window size W
static constexpr int kHD = 64;      // head dim
static constexpr int kNW = 64;      // number of windows
static constexpr int k3D = 3072;    // 3*D

typedef __bf16 v16bf __attribute__((ext_vector_type(16)));
typedef float  v8f   __attribute__((ext_vector_type(8)));

__device__ __forceinline__ v8f wmma_bf16(v16bf a, v16bf b, v8f c) {
    // D = A(16x32 bf16) * B(32x16 bf16) + C(16x16 f32)
    return __builtin_amdgcn_wmma_f32_16x16x32_bf16(
        /*neg_a=*/false, a, /*neg_b=*/false, b,
        /*c_mod=*/(short)0, c, /*reuse_a=*/false, /*reuse_b=*/false);
}

// ---------------------------------------------------------------------------
// gfx1250 async global->LDS copy (ASYNCcnt-tracked DMA that bypasses VGPRs).
// Builtin parameter type (from the toolchain's own diagnostic):
//   arg0: int __attribute__((vector_size(16))) addrspace(1)*   (global src)
//   arg1: addrspace(3) counterpart                              (LDS dst)
// ---------------------------------------------------------------------------
#if __has_builtin(__builtin_amdgcn_global_load_async_to_lds_b128)
#define HAVE_ASYNC_LDS 1
#endif

#define GLOBAL_AS __attribute__((address_space(1)))
#define LDS_AS    __attribute__((address_space(3)))

typedef int v4i_vs __attribute__((vector_size(16)));   // GCC-vector int4

__device__ __forceinline__ void async_b128(const __bf16* g, __bf16* l) {
#if defined(HAVE_ASYNC_LDS)
    __builtin_amdgcn_global_load_async_to_lds_b128(
        (v4i_vs GLOBAL_AS*)(uintptr_t)g,
        (v4i_vs LDS_AS*)(uintptr_t)l,
        /*offset=*/0, /*cpol=*/0);
#else
    *(uint4*)l = *(const uint4*)g;
#endif
}

__device__ __forceinline__ void async_wait() {
#if defined(HAVE_ASYNC_LDS)
    asm volatile("s_wait_asynccnt 0x0" ::: "memory");
#endif
}

// ---------------------------------------------------------------------------
// Elementwise helpers
// ---------------------------------------------------------------------------
__global__ __launch_bounds__(256) void cast_bf16_k(const float* __restrict__ in,
                                                   __bf16* __restrict__ out, int n) {
    int i = blockIdx.x * 256 + threadIdx.x;
    if (i < n) out[i] = (__bf16)in[i];
}

__global__ __launch_bounds__(256) void zero_f32_k(float* __restrict__ p, int n) {
    int i = blockIdx.x * 256 + threadIdx.x;
    if (i < n) p[i] = 0.0f;
}

// ---------------------------------------------------------------------------
// Tiled bf16 GEMM:  C[M,N] = A[M,K] * Bt[N,K]^T   (A,Bt bf16 row-major)
//   MODE 0: store C as bf16 (qkv projection)
//   MODE 1: epilogue t = tanh(C + bias[col]); atomic row-sum into rowsum[M]
//   MODE 2: store C as f32 with bias (output projection -> d_out)
// Block: 256 threads (8 wave32 waves). Tile 128x128, BK=64, double-buffered
// LDS filled by async global->LDS DMA; 16 WMMAs per wave per K-slab.
// ---------------------------------------------------------------------------
template <int MODE>
__global__ __launch_bounds__(256) void gemm_bf16_k(
    const __bf16* __restrict__ A, const __bf16* __restrict__ Bt,
    void* __restrict__ Cout, const float* __restrict__ bias,
    float* __restrict__ rowsum, int M, int N, int K)
{
    __shared__ __bf16 As[2][128 * 64];   // 16 KB per buffer
    __shared__ __bf16 Bs[2][128 * 64];

    const int tid  = threadIdx.x;
    const int m0   = blockIdx.y * 128;
    const int n0   = blockIdx.x * 128;
    const int lane = tid & 31;
    const int wid  = tid >> 5;
    const int wm   = wid & 3;        // 4 M-subtiles of 32 rows
    const int wn   = wid >> 2;       // 2 N-subtiles of 64 cols
    const int mr   = lane & 15;
    const int kh   = (lane >> 4) * 16;

    const v8f zero = {};
    v8f acc[2][4];
#pragma unroll
    for (int i = 0; i < 2; ++i)
#pragma unroll
        for (int j = 0; j < 4; ++j) acc[i][j] = zero;

    // Staging: each thread moves 64 B (4 x b128) per matrix per K-slab.
    const int srow  = tid >> 1;          // 128 rows, 2 threads per row
    const int shalf = (tid & 1) * 32;    // 32 bf16 halves of a 64-wide slab

    auto stage = [&](int buf, int k0) {
        const __bf16* ga = A  + (size_t)(m0 + srow) * K + k0 + shalf;
        const __bf16* gb = Bt + (size_t)(n0 + srow) * K + k0 + shalf;
        __bf16* la = &As[buf][srow * 64 + shalf];
        __bf16* lb = &Bs[buf][srow * 64 + shalf];
#pragma unroll
        for (int c = 0; c < 4; ++c) {
            async_b128(ga + c * 8, la + c * 8);
            async_b128(gb + c * 8, lb + c * 8);
        }
        if (k0 + 64 < K) {               // pull the slab after next toward L2
            __builtin_prefetch(ga + 64, 0, 1);
            __builtin_prefetch(gb + 64, 0, 1);
        }
    };

    const int nk = K / 64;
    stage(0, 0);
    async_wait();
    __syncthreads();

    for (int it = 0; it < nk; ++it) {
        const int buf = it & 1;
        if (it + 1 < nk) stage(buf ^ 1, (it + 1) * 64);  // DMA next slab

        // 16 WMMAs on the resident slab while the DMA runs.
#pragma unroll
        for (int ks = 0; ks < 64; ks += 32) {
            v16bf af[2], bfrag[4];
#pragma unroll
            for (int i = 0; i < 2; ++i)
                af[i] = *(const v16bf*)&As[buf][(wm * 32 + i * 16 + mr) * 64 + ks + kh];
#pragma unroll
            for (int j = 0; j < 4; ++j)
                bfrag[j] = *(const v16bf*)&Bs[buf][(wn * 64 + j * 16 + mr) * 64 + ks + kh];
#pragma unroll
            for (int i = 0; i < 2; ++i)
#pragma unroll
                for (int j = 0; j < 4; ++j)
                    acc[i][j] = wmma_bf16(af[i], bfrag[j], acc[i][j]);
        }

        async_wait();        // own async loads landed in LDS
        __syncthreads();     // every wave's loads landed
    }

    // Epilogue. C/D layout: VGPR v, lanes 0-15 -> row v, lanes 16-31 -> row 8+v.
#pragma unroll
    for (int i = 0; i < 2; ++i) {
#pragma unroll
        for (int j = 0; j < 4; ++j) {
#pragma unroll
            for (int v = 0; v < 8; ++v) {
                const int gr = m0 + wm * 32 + i * 16 + (lane >> 4) * 8 + v;
                const int gc = n0 + wn * 64 + j * 16 + (lane & 15);
                float val = acc[i][j][v];
                if (MODE == 0) {
                    ((__bf16*)Cout)[(size_t)gr * N + gc] = (__bf16)val;
                } else if (MODE == 2) {
                    ((float*)Cout)[(size_t)gr * N + gc] = val + bias[gc];
                } else {
                    // fused tanh + row reduction across the 16 lanes sharing a row
                    float t = tanhf(val + bias[gc]);
                    t += __shfl_xor(t, 1, 32);
                    t += __shfl_xor(t, 2, 32);
                    t += __shfl_xor(t, 4, 32);
                    t += __shfl_xor(t, 8, 32);
                    if ((lane & 15) == 0) atomicAdd(&rowsum[gr], t);
                }
            }
        }
    }
}

// ---------------------------------------------------------------------------
// Windowed attention: one block per (b, h, n) window.
// S = (Q K^T) * 0.125 -> softmax rows -> O = P V, all via bf16 WMMA.
// Q/K/V window tiles staged by async global->LDS DMA.
// ---------------------------------------------------------------------------
__global__ __launch_bounds__(256) void win_attn_k(const __bf16* __restrict__ qkvb,
                                                  float* __restrict__ localf)
{
    const int blk = blockIdx.x;
    const int n = blk & 63;
    const int h = (blk >> 6) & 15;
    const int b = blk >> 10;

    __shared__ __bf16 Qs[64 * 64];   // Q, later reused for P (bf16 probs)
    __shared__ __bf16 Ks[64 * 64];
    __shared__ __bf16 Vs[64 * 64];
    __shared__ float  Ss[64 * 64];   // scores f32

    const int tid = threadIdx.x;
    const size_t base = ((size_t)b * kT + (size_t)n * kWin) * k3D + (size_t)h * kHD;

    // Stage Q/K/V window tiles (64x64 bf16 each) into LDS via async DMA.
    {
        const int row   = tid >> 2;       // 64 rows
        const int chunk = (tid & 3) * 16; // 4 chunks of 16 bf16
        const __bf16* g = qkvb + base + (size_t)row * k3D + chunk;
        __bf16* lq = &Qs[row * 64 + chunk];
        __bf16* lk = &Ks[row * 64 + chunk];
        __bf16* lv = &Vs[row * 64 + chunk];
        async_b128(g,          lq);
        async_b128(g + 8,      lq + 8);
        async_b128(g + kD,     lk);
        async_b128(g + kD + 8, lk + 8);
        async_b128(g + 2*kD,     lv);
        async_b128(g + 2*kD + 8, lv + 8);
    }
    async_wait();
    __syncthreads();

    const int lane = tid & 31;
    const int wid  = tid >> 5;
    const int ti   = wid >> 1;            // S row tile 0..3
    const int tj0  = (wid & 1) * 2;       // two col tiles per wave
    const int mr   = lane & 15;
    const int kh   = (lane >> 4) * 16;
    const v8f zero = {};

    // ---- S = Q K^T * scale -------------------------------------------------
#pragma unroll
    for (int jj = 0; jj < 2; ++jj) {
        const int tj = tj0 + jj;
        v8f sacc = zero;
#pragma unroll
        for (int k0 = 0; k0 < 64; k0 += 32) {
            v16bf a  = *(const v16bf*)&Qs[(ti * 16 + mr) * 64 + k0 + kh];
            v16bf kb = *(const v16bf*)&Ks[(tj * 16 + mr) * 64 + k0 + kh];
            sacc = wmma_bf16(a, kb, sacc);
        }
#pragma unroll
        for (int v = 0; v < 8; ++v) {
            Ss[(ti * 16 + (lane >> 4) * 8 + v) * 64 + tj * 16 + mr] = sacc[v] * 0.125f;
        }
    }
    __syncthreads();

    // ---- row softmax (64 rows, one per thread), P (bf16) overwrites Qs ----
    if (tid < 64) {
        float mx = -1e30f;
        for (int w = 0; w < 64; ++w) mx = fmaxf(mx, Ss[tid * 64 + w]);
        float s = 0.0f;
        for (int w = 0; w < 64; ++w) {
            float e = __expf(Ss[tid * 64 + w] - mx);
            Ss[tid * 64 + w] = e;
            s += e;
        }
        const float inv = 1.0f / s;
        for (int w = 0; w < 64; ++w)
            Qs[tid * 64 + w] = (__bf16)(Ss[tid * 64 + w] * inv);
    }
    __syncthreads();

    // ---- O = P V ----------------------------------------------------------
#pragma unroll
    for (int jj = 0; jj < 2; ++jj) {
        const int tj = tj0 + jj;
        v8f oacc = zero;
#pragma unroll
        for (int k0 = 0; k0 < 64; k0 += 32) {
            v16bf p  = *(const v16bf*)&Qs[(ti * 16 + mr) * 64 + k0 + kh];
            v16bf vb;
#pragma unroll
            for (int s = 0; s < 16; ++s) vb[s] = Vs[(k0 + kh + s) * 64 + tj * 16 + mr];
            oacc = wmma_bf16(p, vb, oacc);
        }
#pragma unroll
        for (int v = 0; v < 8; ++v) {
            const int r = ti * 16 + (lane >> 4) * 8 + v;
            const int c = tj * 16 + mr;
            localf[((size_t)b * kT + (size_t)n * kWin + r) * kD + (size_t)h * kHD + c] = oacc[v];
        }
    }
}

// ---------------------------------------------------------------------------
// Window score softmax + weighted summaries: one block per (b, n).
// rowsum holds sum over D of tanh(...); mean = rowsum / D.
// ---------------------------------------------------------------------------
__global__ __launch_bounds__(256) void win_summary_k(const float* __restrict__ x,
                                                     const float* __restrict__ rowsum,
                                                     float* __restrict__ summ)
{
    const int blk = blockIdx.x;
    const int n = blk & 63;
    const int b = blk >> 6;
    const int tid = threadIdx.x;

    __shared__ float wts[64];
    if (tid < 64) wts[tid] = rowsum[b * kT + n * kWin + tid] * (1.0f / (float)kD);
    __syncthreads();

    float mx = -1e30f;
    for (int w = 0; w < 64; ++w) mx = fmaxf(mx, wts[w]);
    float s = 0.0f;
    for (int w = 0; w < 64; ++w) s += __expf(wts[w] - mx);
    const float inv = 1.0f / s;
    __syncthreads();
    if (tid < 64) wts[tid] = __expf(wts[tid] - mx) * inv;
    __syncthreads();

#pragma unroll
    for (int c = 0; c < 4; ++c) {
        const int d = tid + c * 256;
        float a = 0.0f;
        for (int w = 0; w < 64; ++w)
            a += wts[w] * x[((size_t)b * kT + (size_t)n * kWin + w) * kD + d];
        summ[((size_t)b * kNW + n) * kD + d] = a;
    }
}

// ---------------------------------------------------------------------------
// Cross-window summary attention (tiny): one block per batch.
// ---------------------------------------------------------------------------
__global__ __launch_bounds__(256) void cluster_attn_k(const float* __restrict__ summ,
                                                      const int* __restrict__ jitter,
                                                      float* __restrict__ cs)
{
    const int b = blockIdx.x;
    const int tid = threadIdx.x;

    __shared__ float part[64][4];
    __shared__ float dots[64];
    __shared__ float prob[64];
    __shared__ int   jit[64];

    if (tid < 64) jit[tid] = jitter[tid];
    __syncthreads();

    for (int n = 0; n < 64; ++n) {
        const int m  = tid & 63;
        const int ch = tid >> 6;     // 4 chunks of 256 dims
        const float* sn = summ + ((size_t)b * kNW + n) * kD;
        const float* sm = summ + ((size_t)b * kNW + jit[m]) * kD;
        float p = 0.0f;
        for (int d = ch * 256; d < ch * 256 + 256; ++d) p += sn[d] * sm[d];
        part[m][ch] = p;
        __syncthreads();

        if (tid < 64)
            dots[tid] = (part[tid][0] + part[tid][1] + part[tid][2] + part[tid][3]) * 0.03125f; // D^-0.5
        __syncthreads();

        float mx = -1e30f;
        for (int mm = 0; mm < 64; ++mm) mx = fmaxf(mx, dots[mm]);
        float s = 0.0f;
        for (int mm = 0; mm < 64; ++mm) s += __expf(dots[mm] - mx);
        if (tid < 64) prob[tid] = __expf(dots[tid] - mx) / s;
        __syncthreads();

#pragma unroll
        for (int c = 0; c < 4; ++c) {
            const int d = tid + c * 256;
            float a = 0.0f;
            for (int mm = 0; mm < 64; ++mm)
                a += prob[mm] * summ[((size_t)b * kNW + jit[mm]) * kD + d];
            cs[((size_t)b * kNW + n) * kD + d] = a;
        }
        __syncthreads();
    }
}

// ---------------------------------------------------------------------------
// final = local + 0.25 * cs_expanded, cast to bf16 for the output GEMM.
// ---------------------------------------------------------------------------
__global__ __launch_bounds__(256) void combine_k(const float* __restrict__ localf,
                                                 const float* __restrict__ cs,
                                                 __bf16* __restrict__ finalb)
{
    const size_t i = (size_t)blockIdx.x * 256 + threadIdx.x;
    const size_t b = i >> 22;                      // T*D = 2^22
    const size_t r = i & ((size_t(1) << 22) - 1);
    const int t = (int)(r >> 10);
    const int d = (int)(r & 1023);
    const int n = t >> 6;
    const float v = localf[i] + 0.25f * cs[(b * kNW + n) * kD + d];
    finalb[i] = (__bf16)v;
}

// ---------------------------------------------------------------------------
// Host-side launch
// ---------------------------------------------------------------------------
extern "C" void kernel_launch(void* const* d_in, const int* in_sizes, int n_in,
                              void* d_out, int out_size, void* d_ws, size_t ws_size,
                              hipStream_t stream)
{
    (void)in_sizes; (void)n_in; (void)out_size; (void)ws_size;

    const float* x       = (const float*)d_in[0];
    const float* W_qkv   = (const float*)d_in[1];
    const float* W_out   = (const float*)d_in[2];
    const float* b_out   = (const float*)d_in[3];
    const float* W_cross = (const float*)d_in[4];
    const float* b_cross = (const float*)d_in[5];
    const int*   jitter  = (const int*)d_in[6];

    char* w = (char*)d_ws;
    auto take = [&](size_t bytes) {
        void* p = (void*)w;
        w += (bytes + 255) & ~size_t(255);
        return p;
    };
    __bf16* xb      = (__bf16*)take((size_t)kB * kT * kD * 2);
    __bf16* wqkvb   = (__bf16*)take((size_t)k3D * kD * 2);
    __bf16* wcrossb = (__bf16*)take((size_t)kD * kD * 2);
    __bf16* woutb   = (__bf16*)take((size_t)kD * kD * 2);
    __bf16* qkvb    = (__bf16*)take((size_t)kB * kT * k3D * 2);
    float*  localf  = (float*)take((size_t)kB * kT * kD * 4);
    float*  scores  = (float*)take((size_t)kB * kT * 4);
    float*  summ    = (float*)take((size_t)kB * kNW * kD * 4);
    float*  cs      = (float*)take((size_t)kB * kNW * kD * 4);
    __bf16* finalb  = (__bf16*)take((size_t)kB * kT * kD * 2);

    const int nx = kB * kT * kD;             // 16,777,216
    const dim3 blk(256);

    cast_bf16_k<<<(nx + 255) / 256, blk, 0, stream>>>(x, xb, nx);
    cast_bf16_k<<<(k3D * kD + 255) / 256, blk, 0, stream>>>(W_qkv, wqkvb, k3D * kD);
    cast_bf16_k<<<(kD * kD + 255) / 256, blk, 0, stream>>>(W_cross, wcrossb, kD * kD);
    cast_bf16_k<<<(kD * kD + 255) / 256, blk, 0, stream>>>(W_out, woutb, kD * kD);
    zero_f32_k<<<(kB * kT + 255) / 256, blk, 0, stream>>>(scores, kB * kT);

    // qkv = x @ W_qkv^T  (bf16 WMMA, bf16 out)
    gemm_bf16_k<0><<<dim3(k3D / 128, (kB * kT) / 128), blk, 0, stream>>>(
        xb, wqkvb, qkvb, nullptr, nullptr, kB * kT, k3D, kD);

    // windowed attention -> localf (f32)
    win_attn_k<<<kB * kH * kNW, blk, 0, stream>>>(qkvb, localf);

    // scores = rowsum(tanh(x @ W_cross^T + b_cross)) fused into epilogue
    gemm_bf16_k<1><<<dim3(kD / 128, (kB * kT) / 128), blk, 0, stream>>>(
        xb, wcrossb, nullptr, b_cross, scores, kB * kT, kD, kD);

    win_summary_k<<<kB * kNW, blk, 0, stream>>>(x, scores, summ);
    cluster_attn_k<<<kB, blk, 0, stream>>>(summ, jitter, cs);
    combine_k<<<nx / 256, blk, 0, stream>>>(localf, cs, finalb);

    // out = final @ W_out^T + b_out  (f32 out -> d_out)
    gemm_bf16_k<2><<<dim3(kD / 128, (kB * kT) / 128), blk, 0, stream>>>(
        finalb, woutb, d_out, b_out, nullptr, kB * kT, kD, kD);
}
